// AgentAttentionBlock_22909355557405
// MI455X (gfx1250) — matrix operand order repaired and checked by reading, not compiled, not run on hardware
//
#include <hip/hip_runtime.h>

#define DIM   256
#define NHEAD 8
#define HD    32
#define AGN   49
#define APAD  64
#define BB    8
#define NN    4096
#define NTOK  (BB * NN)   // 32768

__constant__ constexpr float SCALE_F = 0.17677669529663687f; // 32^-0.5

typedef __bf16 bf_t;
typedef __attribute__((ext_vector_type(16))) __bf16 bf16x16;
typedef __attribute__((ext_vector_type(8)))  __bf16 bf16x8;
typedef __attribute__((ext_vector_type(8)))  float  f32x8;

__device__ __forceinline__ f32x8 wmma_bf16(bf16x16 a, bf16x16 b, f32x8 c) {
  return __builtin_amdgcn_wmma_f32_16x16x32_bf16(
      /*neg_a=*/false, a, /*neg_b=*/false, b,
      /*c_mod=*/(short)0, c, /*reuse_a=*/false, /*reuse_b=*/false);
}

// A operand (16x32 bf16) from row-major [M][K] tile; lane<16 holds K {0..7,16..23}
__device__ __forceinline__ bf16x16 load_A_rm(const bf_t* tile, int ld, int lane) {
  const int r  = lane & 15;
  const int hk = (lane >> 4) * 8;
  const bf_t* rp = tile + (size_t)r * ld + hk;
  bf16x8 lo = *(const bf16x8*)(rp);
  bf16x8 hi = *(const bf16x8*)(rp + 16);
  bf16x16 o;
#pragma unroll
  for (int i = 0; i < 8; ++i) { o[i] = lo[i]; o[i + 8] = hi[i]; }
  return o;
}

// B operand (32x16 bf16) from B^T row-major [N][K] tile; lane<16 holds K {0..7,16..23} of col N=lane
__device__ __forceinline__ bf16x16 load_B_t(const bf_t* tile, int ld, int lane) {
  const int n  = lane & 15;
  const int hk = (lane >> 4) * 8;
  const bf_t* rp = tile + (size_t)n * ld + hk;
  bf16x8 lo = *(const bf16x8*)(rp);
  bf16x8 hi = *(const bf16x8*)(rp + 16);
  bf16x16 o;
#pragma unroll
  for (int i = 0; i < 8; ++i) { o[i] = lo[i]; o[i + 8] = hi[i]; }
  return o;
}

// ---------------- conversion kernels ----------------
__global__ void k_conv_x(const float* __restrict__ x, bf_t* __restrict__ xb, int n) {
  int i = blockIdx.x * blockDim.x + threadIdx.x;
  int stride = gridDim.x * blockDim.x;
  for (; i < n; i += stride) xb[i] = (bf_t)x[i];
}

__global__ void k_conv_w(const float* __restrict__ Wq, const float* __restrict__ Wkv,
                         const float* __restrict__ agent, const float* __restrict__ Wproj,
                         bf_t* __restrict__ WqkvT, bf_t* __restrict__ WprojT,
                         bf_t* __restrict__ agb) {
  const int nq = 768 * 256, np = 256 * 256, na = NHEAD * APAD * HD;
  int i = blockIdx.x * blockDim.x + threadIdx.x;
  int stride = gridDim.x * blockDim.x;
  for (int t = i; t < nq + np + na; t += stride) {
    if (t < nq) {                       // WqkvT[j][k] = W[k][j], j in 0..767
      int j = t >> 8, k = t & 255;
      float v = (j < 256) ? Wq[k * 256 + j] : Wkv[k * 512 + (j - 256)];
      WqkvT[t] = (bf_t)v;
    } else if (t < nq + np) {           // WprojT[j][k] = Wproj[k][j]
      int u = t - nq;
      int j = u >> 8, k = u & 255;
      WprojT[u] = (bf_t)Wproj[k * 256 + j];
    } else {                            // agb[h][a(pad 64)][d]
      int u = t - nq - np;
      int h = u / (APAD * HD);
      int a = (u / HD) % APAD;
      int d = u % HD;
      float v = (a < AGN) ? agent[a * DIM + h * HD + d] : 0.0f;
      agb[u] = (bf_t)v;
    }
  }
}

// ---------------- fused QKV projection GEMM: [32768,256] x [256,768] ----------------
__global__ void __launch_bounds__(256) k_qkv(const bf_t* __restrict__ xb,
                                             const bf_t* __restrict__ WqkvT,
                                             const float* __restrict__ bq,
                                             const float* __restrict__ bkv,
                                             bf_t* __restrict__ qh, bf_t* __restrict__ kh,
                                             bf_t* __restrict__ vT) {
  const int lane = threadIdx.x & 31;
  const int wv   = threadIdx.x >> 5;
  const int tile = blockIdx.x * 8 + wv;      // 512 x 12 tiles of 64x64
  const int m0 = (tile / 12) * 64;
  const int n0 = (tile % 12) * 64;
  f32x8 zero = {};
  f32x8 C[4][4];
#pragma unroll
  for (int i = 0; i < 4; ++i)
#pragma unroll
    for (int j = 0; j < 4; ++j) C[i][j] = zero;

  for (int kk = 0; kk < 256; kk += 32) {
    __builtin_prefetch(xb + (size_t)m0 * 256 + kk + 32, 0, 1);
    bf16x16 A[4], Bm[4];
#pragma unroll
    for (int i = 0; i < 4; ++i) A[i]  = load_A_rm(xb    + (size_t)(m0 + 16 * i) * 256 + kk, 256, lane);
#pragma unroll
    for (int j = 0; j < 4; ++j) Bm[j] = load_B_t(WqkvT + (size_t)(n0 + 16 * j) * 256 + kk, 256, lane);
#pragma unroll
    for (int i = 0; i < 4; ++i)
#pragma unroll
      for (int j = 0; j < 4; ++j) C[i][j] = wmma_bf16(A[i], Bm[j], C[i][j]);
  }
  // epilogue: bias + scatter into per-head layouts (q,k: [b,h,n,d]; v: [b,h,d,n] transposed)
#pragma unroll
  for (int i = 0; i < 4; ++i)
#pragma unroll
    for (int j = 0; j < 4; ++j)
#pragma unroll
      for (int v = 0; v < 8; ++v) {
        int row = m0 + 16 * i + (lane < 16 ? v : v + 8);
        int col = n0 + 16 * j + (lane & 15);
        float bias = (col < 256) ? bq[col] : bkv[col - 256];
        float val = C[i][j][v] + bias;
        int b = row >> 12, n = row & 4095;
        if (col < 256) {
          int h = col >> 5, d = col & 31;
          qh[(((size_t)(b * 8 + h)) * 4096 + n) * 32 + d] = (bf_t)val;
        } else if (col < 512) {
          int c = col - 256; int h = c >> 5, d = c & 31;
          kh[(((size_t)(b * 8 + h)) * 4096 + n) * 32 + d] = (bf_t)val;
        } else {
          int c = col - 512; int h = c >> 5, d = c & 31;
          vT[(((size_t)(b * 8 + h)) * 32 + d) * 4096 + n] = (bf_t)val;
        }
      }
}

// ---------------- stage B+C: softmax_n(agent @ k^T) @ v  -> x_aT[b,h,d,a] ----------------
__global__ void __launch_bounds__(128) k_agent_kv(const bf_t* __restrict__ agb,
                                                  const bf_t* __restrict__ kh,
                                                  const bf_t* __restrict__ vT,
                                                  bf_t* __restrict__ xaT) {
  __shared__ __align__(16) float sbuf[4][16][64];
  __shared__ __align__(16) bf_t  pbuf[4][16][64];
  __shared__ float rs[4][16];
  const int lane = threadIdx.x & 31;
  const int wv   = threadIdx.x >> 5;      // wave owns a-rows 16*wv..16*wv+15
  const int bh = blockIdx.x;
  const int h  = bh & 7;
  const bf_t* Kh = kh + (size_t)bh * 4096 * 32;   // [4096][32]
  const bf_t* Vt = vT + (size_t)bh * 32 * 4096;   // [32][4096]
  const bf16x16 Aag = load_A_rm(agb + (size_t)h * APAD * HD + (size_t)(16 * wv) * HD, HD, lane);
  const int r  = lane & 15;
  const int ch = lane >> 4;
  f32x8 zero = {};

  // pass 1: row max of raw scores
  float mx = -3.0e38f;
  for (int t = 0; t < 64; ++t) {
    f32x8 S[4];
#pragma unroll
    for (int j = 0; j < 4; ++j)
      S[j] = wmma_bf16(Aag, load_B_t(Kh + (size_t)(t * 64 + 16 * j) * 32, 32, lane), zero);
#pragma unroll
    for (int j = 0; j < 4; ++j)
#pragma unroll
      for (int v = 0; v < 8; ++v)
        sbuf[wv][(lane < 16 ? v : v + 8)][16 * j + (lane & 15)] = S[j][v];
    __syncthreads();
#pragma unroll
    for (int c = 0; c < 32; ++c) mx = fmaxf(mx, sbuf[wv][r][ch * 32 + c]);
    __syncthreads();
  }
  mx = fmaxf(mx, __shfl_xor(mx, 16, 32));

  // pass 2: exp, accumulate rowsum and P@V
  float rsum = 0.0f;
  f32x8 Cx[2]; Cx[0] = zero; Cx[1] = zero;
  for (int t = 0; t < 64; ++t) {
    f32x8 S[4];
#pragma unroll
    for (int j = 0; j < 4; ++j)
      S[j] = wmma_bf16(Aag, load_B_t(Kh + (size_t)(t * 64 + 16 * j) * 32, 32, lane), zero);
#pragma unroll
    for (int j = 0; j < 4; ++j)
#pragma unroll
      for (int v = 0; v < 8; ++v)
        sbuf[wv][(lane < 16 ? v : v + 8)][16 * j + (lane & 15)] = S[j][v];
    __syncthreads();
#pragma unroll
    for (int c = 0; c < 32; ++c) {
      float e = __expf((sbuf[wv][r][ch * 32 + c] - mx) * SCALE_F);
      rsum += e;
      pbuf[wv][r][ch * 32 + c] = (bf_t)e;
    }
    __syncthreads();
    bf16x16 Ap0 = load_A_rm(&pbuf[wv][0][0],  64, lane);
    bf16x16 Ap1 = load_A_rm(&pbuf[wv][0][32], 64, lane);
#pragma unroll
    for (int jd = 0; jd < 2; ++jd) {
      Cx[jd] = wmma_bf16(Ap0, load_B_t(Vt + (size_t)(16 * jd) * 4096 + t * 64,      4096, lane), Cx[jd]);
      Cx[jd] = wmma_bf16(Ap1, load_B_t(Vt + (size_t)(16 * jd) * 4096 + t * 64 + 32, 4096, lane), Cx[jd]);
    }
    __syncthreads();
  }
  rsum += __shfl_xor(rsum, 16, 32);
  if (lane < 16) rs[wv][r] = rsum;
  __syncthreads();
  // normalize and store x_a transposed: xaT[b,h,d,a]
#pragma unroll
  for (int jd = 0; jd < 2; ++jd)
#pragma unroll
    for (int v = 0; v < 8; ++v) {
      int ar = (lane < 16 ? v : v + 8);
      int a  = 16 * wv + ar;
      int d  = 16 * jd + (lane & 15);
      float val = Cx[jd][v] / rs[wv][ar];
      xaT[((size_t)bh * 32 + d) * APAD + a] = (bf_t)val;
    }
}

// ---------------- stage A+D: softmax_a(q @ agent^T) @ x_a -> out_heads [32768][256] ----------------
__global__ void __launch_bounds__(256) k_agent_q(const bf_t* __restrict__ qh,
                                                 const bf_t* __restrict__ agb,
                                                 const bf_t* __restrict__ xaT,
                                                 bf_t* __restrict__ oh) {
  __shared__ __align__(16) float sbuf[8][16][64];
  __shared__ __align__(16) bf_t  pbuf[8][16][64];
  const int lane = threadIdx.x & 31;
  const int wv   = threadIdx.x >> 5;
  const int tile = blockIdx.x * 8 + wv;     // 64 (b,h) * 256 m-tiles
  const int bh = tile >> 8;
  const int mt = tile & 255;
  const int h  = bh & 7;
  const int m0 = mt * 16;
  f32x8 zero = {};

  bf16x16 Aq = load_A_rm(qh + ((size_t)bh * 4096 + m0) * 32, 32, lane);
  f32x8 S[4];
#pragma unroll
  for (int j = 0; j < 4; ++j)
    S[j] = wmma_bf16(Aq, load_B_t(agb + (size_t)h * APAD * HD + (size_t)(16 * j) * HD, HD, lane), zero);
#pragma unroll
  for (int j = 0; j < 4; ++j)
#pragma unroll
    for (int v = 0; v < 8; ++v)
      sbuf[wv][(lane < 16 ? v : v + 8)][16 * j + (lane & 15)] = S[j][v];
  __syncthreads();

  const int r = lane & 15, ch = lane >> 4;
  float mx = -3.0e38f;
#pragma unroll
  for (int c = 0; c < 32; ++c) {
    int col = ch * 32 + c;
    float s = sbuf[wv][r][col];
    if (col < AGN) mx = fmaxf(mx, s);
  }
  mx = fmaxf(mx, __shfl_xor(mx, 16, 32));
  float rsum = 0.0f;
  float e[32];
#pragma unroll
  for (int c = 0; c < 32; ++c) {
    int col = ch * 32 + c;
    float ev = (col < AGN) ? __expf((sbuf[wv][r][col] - mx) * SCALE_F) : 0.0f;
    e[c] = ev;
    rsum += ev;
  }
  rsum += __shfl_xor(rsum, 16, 32);
  float inv = 1.0f / rsum;
#pragma unroll
  for (int c = 0; c < 32; ++c) pbuf[wv][r][ch * 32 + c] = (bf_t)(e[c] * inv);  // normalized P
  __syncthreads();

  bf16x16 Ap0 = load_A_rm(&pbuf[wv][0][0],  64, lane);
  bf16x16 Ap1 = load_A_rm(&pbuf[wv][0][32], 64, lane);
  f32x8 O[2];
#pragma unroll
  for (int jd = 0; jd < 2; ++jd) {
    O[jd] = wmma_bf16(Ap0, load_B_t(xaT + ((size_t)bh * 32 + 16 * jd) * APAD,      APAD, lane), zero);
    O[jd] = wmma_bf16(Ap1, load_B_t(xaT + ((size_t)bh * 32 + 16 * jd) * APAD + 32, APAD, lane), O[jd]);
  }
  const int b = bh >> 3;
#pragma unroll
  for (int jd = 0; jd < 2; ++jd)
#pragma unroll
    for (int v = 0; v < 8; ++v) {
      int n = m0 + (lane < 16 ? v : v + 8);
      int token = b * 4096 + n;
      int col = h * 32 + 16 * jd + (lane & 15);
      oh[(size_t)token * 256 + col] = (bf_t)O[jd][v];
    }
}

// ---------------- output projection: [32768,256] x [256,256] + bias -> fp32 ----------------
__global__ void __launch_bounds__(256) k_proj(const bf_t* __restrict__ oh,
                                              const bf_t* __restrict__ WprojT,
                                              const float* __restrict__ bproj,
                                              float* __restrict__ out) {
  const int lane = threadIdx.x & 31;
  const int wv   = threadIdx.x >> 5;
  const int tile = blockIdx.x * 8 + wv;      // 512 x 4 tiles of 64x64
  const int m0 = (tile >> 2) * 64;
  const int n0 = (tile & 3) * 64;
  f32x8 zero = {};
  f32x8 C[4][4];
#pragma unroll
  for (int i = 0; i < 4; ++i)
#pragma unroll
    for (int j = 0; j < 4; ++j) C[i][j] = zero;
  for (int kk = 0; kk < 256; kk += 32) {
    __builtin_prefetch(oh + (size_t)m0 * 256 + kk + 32, 0, 1);
    bf16x16 A[4], Bm[4];
#pragma unroll
    for (int i = 0; i < 4; ++i) A[i]  = load_A_rm(oh     + (size_t)(m0 + 16 * i) * 256 + kk, 256, lane);
#pragma unroll
    for (int j = 0; j < 4; ++j) Bm[j] = load_B_t(WprojT + (size_t)(n0 + 16 * j) * 256 + kk, 256, lane);
#pragma unroll
    for (int i = 0; i < 4; ++i)
#pragma unroll
      for (int j = 0; j < 4; ++j) C[i][j] = wmma_bf16(A[i], Bm[j], C[i][j]);
  }
#pragma unroll
  for (int i = 0; i < 4; ++i)
#pragma unroll
    for (int j = 0; j < 4; ++j)
#pragma unroll
      for (int v = 0; v < 8; ++v) {
        int row = m0 + 16 * i + (lane < 16 ? v : v + 8);
        int col = n0 + 16 * j + (lane & 15);
        out[(size_t)row * 256 + col] = C[i][j][v] + bproj[col];
      }
}

extern "C" void kernel_launch(void* const* d_in, const int* in_sizes, int n_in,
                              void* d_out, int out_size, void* d_ws, size_t ws_size,
                              hipStream_t stream) {
  (void)in_sizes; (void)n_in; (void)out_size; (void)ws_size;
  const float* x     = (const float*)d_in[0];
  const float* Wq    = (const float*)d_in[1];
  const float* bq    = (const float*)d_in[2];
  const float* Wkv   = (const float*)d_in[3];
  const float* bkv   = (const float*)d_in[4];
  const float* agent = (const float*)d_in[5];
  const float* Wproj = (const float*)d_in[6];
  const float* bproj = (const float*)d_in[7];
  float* out = (float*)d_out;

  char* ws = (char*)d_ws;
  size_t off = 0;
  auto alloc = [&](size_t bytes) -> void* {
    void* p = ws + off;
    off += (bytes + 255) & ~(size_t)255;
    return p;
  };
  bf_t* xb     = (bf_t*)alloc((size_t)NTOK * DIM * 2);      // 16.8 MB
  bf_t* qh     = (bf_t*)alloc((size_t)64 * 4096 * 32 * 2);  // 16.8 MB
  bf_t* kh     = (bf_t*)alloc((size_t)64 * 4096 * 32 * 2);  // 16.8 MB
  bf_t* vT     = (bf_t*)alloc((size_t)64 * 32 * 4096 * 2);  // 16.8 MB
  bf_t* oh     = (bf_t*)alloc((size_t)NTOK * DIM * 2);      // 16.8 MB
  bf_t* WqkvT  = (bf_t*)alloc((size_t)768 * 256 * 2);
  bf_t* WprojT = (bf_t*)alloc((size_t)256 * 256 * 2);
  bf_t* agb    = (bf_t*)alloc((size_t)NHEAD * APAD * HD * 2);
  bf_t* xaT    = (bf_t*)alloc((size_t)64 * 32 * APAD * 2);

  k_conv_x<<<512, 256, 0, stream>>>(x, xb, NTOK * DIM);
  k_conv_w<<<272, 256, 0, stream>>>(Wq, Wkv, agent, Wproj, WqkvT, WprojT, agb);
  k_qkv<<<768, 256, 0, stream>>>(xb, WqkvT, bq, bkv, qh, kh, vT);       // 6144 wave tiles
  k_agent_kv<<<64, 128, 0, stream>>>(agb, kh, vT, xaT);                 // one WG per (b,h)
  k_agent_q<<<2048, 256, 0, stream>>>(qh, agb, xaT, oh);                // 16384 wave tiles
  k_proj<<<256, 256, 0, stream>>>(oh, WprojT, bproj, out);              // 2048 wave tiles
}
